// net_crossInteraction_17806934409265
// MI455X (gfx1250) — matrix operand order repaired and checked by reading, not compiled, not run on hardware
//
#include <hip/hip_runtime.h>
#include <hip/hip_bf16.h>
#include <stdint.h>

typedef __attribute__((ext_vector_type(16))) __bf16 v16bf;
typedef __attribute__((ext_vector_type(8)))  float  v8f;
typedef __attribute__((ext_vector_type(2)))  float  v2f;

#define DD 256
#define SQ 640
#define INTER_N (SQ * SQ)   // 409600 per batch
#define KK 256              // shared inner dimension of every GEMM in this net

__device__ __forceinline__ float sigf(float x) { return 1.0f / (1.0f + expf(-x)); }

// ---------------------------------------------------------------------------
// bf16 NT GEMM (f32 accumulate), one wave -> 16x64 output tile (4 sub-tiles
// sharing one A fragment). K = KK compile-time -> fully unrolled: 32 WMMAs.
//   C[M,N] = act_c( act_a(A[M,KK]) * Bt[N,KK]^T + bias[N] )
// Fragment layouts (wave32; A/B use contiguous k, C/D per CDNA5 ISA 7.12.2):
//   A (16x32 bf16): lane<16 -> M=lane, e=0..15 -> K=k0+e;
//                   lane>=16 adds +16 to K.
//   B (32x16 bf16): N = lane&15, e -> K = k0 + e + 16*(lane>=16).
//   C/D (8 VGPR f32): N = lane&15, M = r + 8*(lane>=16).
// ---------------------------------------------------------------------------
template<int ACT_A, int ACT_C>
__global__ void gemm_nt_bf16(const float* __restrict__ A, const float* __restrict__ Bt,
                             const float* __restrict__ bias, float* __restrict__ C,
                             int M, int N, int lda, int ldb, int ldc)
{
    const int wave  = (int)((blockIdx.x * blockDim.x + threadIdx.x) >> 5);
    const int lane  = (int)(threadIdx.x & 31);
    const int ncol  = N >> 6;                  // 64-wide N column groups
    const int tiles = (M >> 4) * ncol;
    if (wave >= tiles) return;                 // wave-uniform: EXEC stays all-1s

    const int tm  = wave / ncol;
    const int tn0 = (wave % ncol) << 2;        // first of 4 16-wide N tiles
    const int hf  = (lane >> 4) & 1;
    const int l16 = lane & 15;
    const float* Ao = A + (size_t)(tm * 16 + l16) * lda;

    v8f acc[4] = {};
#pragma unroll
    for (int k0 = 0; k0 < KK; k0 += 32) {
        v16bf a;
#pragma unroll
        for (int e = 0; e < 16; ++e) {
            const int ka = k0 + (hf << 4) + e;
            float av = Ao[ka];
            if (ACT_A) av = av > 0.0f ? av : 0.0f;
            a[e] = (__bf16)av;
        }
#pragma unroll
        for (int j = 0; j < 4; ++j) {
            const float* Bo = Bt + (size_t)((tn0 + j) * 16 + l16) * ldb + k0 + (hf << 4);
            if (k0 + 32 < KK) __builtin_prefetch(Bo + 32, 0, 1);
            v16bf b;
#pragma unroll
            for (int e = 0; e < 16; ++e) b[e] = (__bf16)Bo[e];
            acc[j] = __builtin_amdgcn_wmma_f32_16x16x32_bf16(
                         false, a, false, b, (short)0, acc[j], false, false);
        }
    }

#pragma unroll
    for (int j = 0; j < 4; ++j) {
        const int n  = (tn0 + j) * 16 + l16;
        const float bv = bias ? bias[n] : 0.0f;
#pragma unroll
        for (int r = 0; r < 8; ++r) {
            const int m = tm * 16 + r + (hf << 3);
            float v = acc[j][r] + bv;
            if (ACT_C) v = sigf(v);
            C[(size_t)m * ldc + n] = v;
        }
    }
}

// ---------------------------------------------------------------------------
// Full-precision f32 NT GEMM on V_WMMA_F32_16X16X4_F32 (exact fp32 matmul on
// the matrix pipe) for the precision-sensitive GEMMs: GRU recurrence, q/k,
// and the output-facing Q*K^T. Same 16x64 wave tile.
//   A (16x4 f32): M = lane&15, a[e] = A[m][k0 + 2*(lane>=16) + e]
//   B (4x16 f32): N = lane&15, b[e] = Bt[n][k0 + 2*(lane>=16) + e]
// ---------------------------------------------------------------------------
template<int ACT_A, int ACT_C>
__global__ void gemm_nt_f32(const float* __restrict__ A, const float* __restrict__ Bt,
                            const float* __restrict__ bias, float* __restrict__ C,
                            int M, int N, int lda, int ldb, int ldc)
{
    const int wave  = (int)((blockIdx.x * blockDim.x + threadIdx.x) >> 5);
    const int lane  = (int)(threadIdx.x & 31);
    const int ncol  = N >> 6;
    const int tiles = (M >> 4) * ncol;
    if (wave >= tiles) return;

    const int tm  = wave / ncol;
    const int tn0 = (wave % ncol) << 2;
    const int hf  = (lane >> 4) & 1;
    const int l16 = lane & 15;
    const float* Ao = A + (size_t)(tm * 16 + l16) * lda + (hf << 1);

    v8f acc[4] = {};
#pragma unroll 4
    for (int k0 = 0; k0 < KK; k0 += 4) {
        v2f a;
        float av0 = Ao[k0], av1 = Ao[k0 + 1];
        if (ACT_A) { av0 = av0 > 0.0f ? av0 : 0.0f; av1 = av1 > 0.0f ? av1 : 0.0f; }
        a[0] = av0; a[1] = av1;
#pragma unroll
        for (int j = 0; j < 4; ++j) {
            const float* Bo = Bt + (size_t)((tn0 + j) * 16 + l16) * ldb + (hf << 1) + k0;
            v2f b;
            b[0] = Bo[0]; b[1] = Bo[1];
            acc[j] = __builtin_amdgcn_wmma_f32_16x16x4_f32(
                         false, a, false, b, (short)0, acc[j], false, false);
        }
    }

#pragma unroll
    for (int j = 0; j < 4; ++j) {
        const int n  = (tn0 + j) * 16 + l16;
        const float bv = bias ? bias[n] : 0.0f;
#pragma unroll
        for (int r = 0; r < 8; ++r) {
            const int m = tm * 16 + r + (hf << 3);
            float v = acc[j][r] + bv;
            if (ACT_C) v = sigf(v);
            C[(size_t)m * ldc + n] = v;
        }
    }
}

// ---------------------------------------------------------------------------
__global__ void zero_kernel(float* __restrict__ p, int n)
{
    int i = blockIdx.x * blockDim.x + threadIdx.x;
    if (i < n) p[i] = 0.0f;
}

// Gather embedding rows. Row order = [prot][b*20+i][j] so GRU0 GEMM rows are (seq, t).
__global__ void gather_kernel(const int* __restrict__ p1, const int* __restrict__ p2,
                              const float* __restrict__ emb, float* __restrict__ e)
{
    const int row  = blockIdx.x;           // 0..2559
    const int prot = row / 1280;
    const int loc  = row % 1280;
    const int tok  = (prot ? p2 : p1)[loc];
    e[(size_t)row * DD + threadIdx.x] = emb[(size_t)tok * DD + threadIdx.x];
}

// One GRU step (gates r,z,n in torch order). xg rows are (m, t) with stride T.
__global__ void gru_update(const float* __restrict__ xg, const float* __restrict__ hg,
                           const float* __restrict__ bhh,
                           float* __restrict__ h, float* __restrict__ ys,
                           int B, int T, int t)
{
    const int idx = blockIdx.x * blockDim.x + threadIdx.x;
    if (idx >= B * DD) return;
    const int m = idx >> 8;
    const int d = idx & 255;
    const float* xr  = xg + ((size_t)m * T + t) * 768;
    const float* hgm = hg + (size_t)m * 768;
    const float r = sigf(xr[d]       + hgm[d]       + bhh[d]);
    const float z = sigf(xr[256 + d] + hgm[256 + d] + bhh[256 + d]);
    const float n = tanhf(xr[512 + d] + r * (hgm[512 + d] + bhh[512 + d]));
    const float hn = (1.0f - z) * n + z * h[idx];
    h[idx] = hn;
    ys[((size_t)m * T + t) * DD + d] = hn;
}

// Per-batch sum of inter (for normalization).
__global__ void reduce_inter(const float* __restrict__ inter, float* __restrict__ sums)
{
    __shared__ float red[256];
    const int b   = blockIdx.x / 1600;
    const int off = (blockIdx.x % 1600) * 256 + threadIdx.x;
    float v = (off < INTER_N) ? inter[(size_t)b * INTER_N + off] : 0.0f;
    red[threadIdx.x] = v;
    __syncthreads();
    for (int s = 128; s > 0; s >>= 1) {
        if (threadIdx.x < s) red[threadIdx.x] += red[threadIdx.x + s];
        __syncthreads();
    }
    if (threadIdx.x == 0) atomicAdd(&sums[b], red[0]);
}

__global__ void scale_inter(float* __restrict__ inter, const float* __restrict__ sums)
{
    const int idx = blockIdx.x * blockDim.x + threadIdx.x;
    if (idx >= 2 * INTER_N) return;
    inter[idx] = inter[idx] / sums[idx / INTER_N];
}

// pp[b,d] = sum_{s,t} tanh(e1[b,s,d]*e2[b,t,d]) * inter[b,s,t]
__global__ void pp_kernel(const float* __restrict__ e1, const float* __restrict__ e2,
                          const float* __restrict__ inter, float* __restrict__ pp)
{
    const int b  = blockIdx.x / 40;
    const int s0 = (blockIdx.x % 40) * 16;
    const int d  = threadIdx.x;
    const float* E1 = e1 + (size_t)b * SQ * DD;
    const float* E2 = e2 + (size_t)b * SQ * DD;
    const float* W  = inter + (size_t)b * INTER_N;
    float acc = 0.0f;
    for (int s = s0; s < s0 + 16; ++s) {
        const float a = E1[(size_t)s * DD + d];
        const float* wrow = W + (size_t)s * SQ;
        for (int t = 0; t < SQ; ++t) {
            acc += tanhf(a * E2[(size_t)t * DD + d]) * wrow[t];
        }
    }
    atomicAdd(&pp[b * DD + d], acc);
}

// conv1d(k=4,s=2,pad=1) + leaky + maxpool(4) + 3 FC layers; one block per batch.
__global__ void head_kernel(const float* __restrict__ pp,
                            const float* __restrict__ cw, const float* __restrict__ cb,
                            const float* __restrict__ r1w, const float* __restrict__ r1b,
                            const float* __restrict__ r2w, const float* __restrict__ r2b,
                            const float* __restrict__ r3w, const float* __restrict__ r3b,
                            float* __restrict__ affn)
{
    __shared__ float s_pp[DD];
    __shared__ float s_pool[2048];
    __shared__ float s_x1[600];
    __shared__ float s_x2[300];
    __shared__ float s_red[256];
    const int b = blockIdx.x, tid = threadIdx.x;
    s_pp[tid] = pp[b * DD + tid];
    __syncthreads();

    for (int p = tid; p < 2048; p += 256) {      // p = c*32 + pw
        const int c = p >> 5, pw = p & 31;
        float mx = -3.4e38f;
        for (int q = 0; q < 4; ++q) {
            const int o = pw * 4 + q;            // conv output pos 0..127
            float a = cb[c];
            for (int kk = 0; kk < 4; ++kk) {
                const int ii = 2 * o - 1 + kk;
                const float xv = (ii >= 0 && ii < DD) ? s_pp[ii] : 0.0f;
                a += xv * cw[c * 4 + kk];
            }
            a = a > 0.0f ? a : 0.1f * a;
            mx = fmaxf(mx, a);
        }
        s_pool[p] = mx;
    }
    __syncthreads();

    for (int row = tid; row < 600; row += 256) {
        float a = r1b[row];
        const float* wr = r1w + (size_t)row * 2048;
        for (int i = 0; i < 2048; ++i) a += wr[i] * s_pool[i];
        s_x1[row] = a > 0.0f ? a : 0.1f * a;
    }
    __syncthreads();

    for (int row = tid; row < 300; row += 256) {
        float a = r2b[row];
        const float* wr = r2w + (size_t)row * 600;
        for (int i = 0; i < 600; ++i) a += wr[i] * s_x1[i];
        s_x2[row] = a > 0.0f ? a : 0.1f * a;
    }
    __syncthreads();

    float part = 0.0f;
    for (int i = tid; i < 300; i += 256) part += s_x2[i] * r3w[i];
    s_red[tid] = part;
    __syncthreads();
    for (int s = 128; s > 0; s >>= 1) {
        if (tid < s) s_red[tid] += s_red[tid + s];
        __syncthreads();
    }
    if (tid == 0) affn[b] = s_red[0] + r3b[0];
}

// ---------------------------------------------------------------------------
static inline int gemm_blocks(int M, int N)   // waves = (M/16)*(N/64), 8 waves/block
{
    const int tiles = (M / 16) * (N / 64);
    return (tiles + 7) / 8;
}

static inline void launch_bf16(hipStream_t s, const float* A, const float* Bt,
                               const float* bias, float* C, int M, int N,
                               int lda, int ldb, int ldc)
{
    gemm_nt_bf16<0, 0><<<gemm_blocks(M, N), 256, 0, s>>>(A, Bt, bias, C, M, N, lda, ldb, ldc);
}

static inline void launch_f32(hipStream_t s, const float* A, const float* Bt,
                              const float* bias, float* C, int M, int N,
                              int lda, int ldb, int ldc, int act_a, int act_c)
{
    const int blocks = gemm_blocks(M, N);
    if (act_a == 1)
        gemm_nt_f32<1, 0><<<blocks, 256, 0, s>>>(A, Bt, bias, C, M, N, lda, ldb, ldc);
    else if (act_c == 1)
        gemm_nt_f32<0, 1><<<blocks, 256, 0, s>>>(A, Bt, bias, C, M, N, lda, ldb, ldc);
    else
        gemm_nt_f32<0, 0><<<blocks, 256, 0, s>>>(A, Bt, bias, C, M, N, lda, ldb, ldc);
}

static inline void launch_zero(hipStream_t s, float* p, int n)
{
    zero_kernel<<<(n + 255) / 256, 256, 0, s>>>(p, n);
}

extern "C" void kernel_launch(void* const* d_in, const int* in_sizes, int n_in,
                              void* d_out, int out_size, void* d_ws, size_t ws_size,
                              hipStream_t stream)
{
    (void)in_sizes; (void)n_in; (void)out_size; (void)ws_size;

    const int*   p1     = (const int*)  d_in[0];
    const int*   p2     = (const int*)  d_in[1];
    const float* emb    = (const float*)d_in[2];
    const float* g0_wih = (const float*)d_in[3];
    const float* g0_whh = (const float*)d_in[4];
    const float* g0_bih = (const float*)d_in[5];
    const float* g0_bhh = (const float*)d_in[6];
    const float* g1_wih = (const float*)d_in[7];
    const float* g1_whh = (const float*)d_in[8];
    const float* g1_bih = (const float*)d_in[9];
    const float* g1_bhh = (const float*)d_in[10];
    const float* ja1_w  = (const float*)d_in[11];
    const float* ja1_b  = (const float*)d_in[12];
    const float* ja2_w  = (const float*)d_in[13];
    const float* ja2_b  = (const float*)d_in[14];
    const float* conv_w = (const float*)d_in[15];
    const float* conv_b = (const float*)d_in[16];
    const float* r1_w   = (const float*)d_in[17];
    const float* r1_b   = (const float*)d_in[18];
    const float* r2_w   = (const float*)d_in[19];
    const float* r2_b   = (const float*)d_in[20];
    const float* r3_w   = (const float*)d_in[21];
    const float* r3_b   = (const float*)d_in[22];

    float* out = (float*)d_out;          // inter [2*640*640] then affn [2]
    float* ws  = (float*)d_ws;

    // workspace layout (floats), ~18.9 MB total
    float* e    = ws;                    // 2560*256
    float* xg   = e    + 2560 * 256;     // 2560*768 (reused for both GRU stages)
    float* ys0  = xg   + 2560 * 768;     // 2560*256
    float* efin = ys0  + 2560 * 256;     // 2560*256
    float* qb   = efin + 2560 * 256;     // 1280*256
    float* kb   = qb   + 1280 * 256;     // 1280*256
    float* h    = kb   + 1280 * 256;     // 128*256
    float* hg   = h    + 128 * 256;      // 128*768
    float* sums = hg   + 128 * 768;      // 2
    float* pp   = sums + 2;              // 512

    // 1) embedding gather: rows (prot, b*20+i, j)
    gather_kernel<<<2560, 256, 0, stream>>>(p1, p2, emb, e);

    // 2) GRU0: xg = e @ w_ih^T + b_ih (bf16 WMMA) ; 32 recurrent steps (f32 WMMA, B=80)
    launch_bf16(stream, e, g0_wih, g0_bih, xg, 2560, 768, 256, 256, 768);
    launch_zero(stream, h, 80 * 256);
    for (int t = 0; t < 32; ++t) {
        launch_f32(stream, h, g0_whh, nullptr, hg, 80, 768, 256, 256, 768, 0, 0);
        gru_update<<<(80 * 256 + 255) / 256, 256, 0, stream>>>(xg, hg, g0_bhh, h, ys0, 80, 32, t);
    }

    // 3) GRU1: flat reshape is free; 20 steps (B=128)
    launch_bf16(stream, ys0, g1_wih, g1_bih, xg, 2560, 768, 256, 256, 768);
    launch_zero(stream, h, 128 * 256);
    for (int t = 0; t < 20; ++t) {
        launch_f32(stream, h, g1_whh, nullptr, hg, 128, 768, 256, 256, 768, 0, 0);
        gru_update<<<(128 * 256 + 255) / 256, 256, 0, stream>>>(xg, hg, g1_bhh, h, efin, 128, 20, t);
    }

    // 4) q = relu(e1)@ja1^T + b ; k = relu(e2)@ja2^T + b   (f32 WMMA, relu fused on A)
    launch_f32(stream, efin,              ja1_w, ja1_b, qb, 1280, 256, 256, 256, 256, 1, 0);
    launch_f32(stream, efin + 1280 * 256, ja2_w, ja2_b, kb, 1280, 256, 256, 256, 256, 1, 0);

    // 5) inter = sigmoid(Q K^T) per batch, straight into d_out (f32 WMMA, sigmoid fused)
    for (int b = 0; b < 2; ++b) {
        launch_f32(stream, qb + (size_t)b * SQ * DD, kb + (size_t)b * SQ * DD, nullptr,
                   out + (size_t)b * INTER_N, SQ, SQ, 256, 256, SQ, 0, 1);
    }

    // 6) normalize by per-batch sum
    launch_zero(stream, sums, 2);
    reduce_inter<<<3200, 256, 0, stream>>>(out, sums);
    scale_inter<<<(2 * INTER_N + 255) / 256, 256, 0, stream>>>(out, sums);

    // 7) pp[b,d] = sum_{s,t} tanh(e1*e2) * inter
    launch_zero(stream, pp, 512);
    pp_kernel<<<80, 256, 0, stream>>>(efin, efin + 1280 * 256, out, pp);

    // 8) conv + pool + MLP head -> affn
    head_kernel<<<2, 256, 0, stream>>>(pp, conv_w, conv_b, r1_w, r1_b, r2_w, r2_b,
                                       r3_w, r3_b, out + 2 * INTER_N);
}
